// MMAB_33784212750681
// MI455X (gfx1250) — matrix operand order repaired and checked by reading, not compiled, not run on hardware
//
#include <hip/hip_runtime.h>
#include <math.h>

typedef __attribute__((ext_vector_type(16))) _Float16 v16h;
typedef __attribute__((ext_vector_type(8)))  _Float16 h8;
typedef __attribute__((ext_vector_type(8)))  float    v8f;
typedef __attribute__((ext_vector_type(4)))  float    f4;
typedef __attribute__((ext_vector_type(4)))  _Float16 h4;

#define DEV static __device__ __forceinline__

static constexpr int B_  = 8;
static constexpr int N_  = 1024;   // Nq == Nk
static constexpr int D_  = 512;
static constexpr int H_  = 8;
static constexpr int DH_ = 64;     // head dim
#define SCALE 0.044194173824159216f   /* 1/sqrt(512) */

DEV int laneid() { return threadIdx.x & 31; }

DEV v8f wmma(v16h a, v16h b, v8f c) {
  return __builtin_amdgcn_wmma_f32_16x16x32_f16(false, a, false, b, (short)0, c, false, false);
}

// async copy of 16 bytes/lane: global -> LDS (tracked by ASYNCcnt)
DEV void async_b128(unsigned lds_addr, const void* gaddr) {
  asm volatile("global_load_async_to_lds_b128 %0, %1, off"
               :: "v"(lds_addr), "v"(gaddr) : "memory");
}
DEV unsigned lds_off(const void* p) { return (unsigned)(uintptr_t)p; }

// A-fragment 16x32 f16 from row-major src (ld in elements), k-window [0,32).
// lanes 0-15 (M=0..15): halfs 0..7 = K 0..7, halfs 8..15 = K 16..23
// lanes 16-31:          halfs 0..7 = K 8..15, halfs 8..15 = K 24..31
DEV v16h load_fragA(const _Float16* src, int ld) {
  int l = laneid();
  int m = l & 15, hi = (l >> 4) << 3;
  const _Float16* p = src + (size_t)m * ld + hi;
  h8 a = *(const h8*)(p);
  h8 b = *(const h8*)(p + 16);
  return __builtin_shufflevector(a, b, 0,1,2,3,4,5,6,7,8,9,10,11,12,13,14,15);
}

// B-fragment 32x16 f16 (K x N) from row-major *transposed* buffer BT (row n = B[:,n]).
// lanes 0-15 (N=0..15): halfs 0..7 = K 0..7, halfs 8..15 = K 16..23; lanes 16-31: K 8..15, K 24..31
DEV v16h load_fragB(const _Float16* srcT, int ld) {
  int l = laneid();
  int n = l & 15, hi = (l >> 4) << 3;
  const _Float16* p = srcT + (size_t)n * ld + hi;
  h8 a = *(const h8*)(p);
  h8 b = *(const h8*)(p + 16);
  return __builtin_shufflevector(a, b, 0,1,2,3,4,5,6,7,8,9,10,11,12,13,14,15);
}

// ---------------- prep kernels ----------------
__global__ void k_cvt16(const float* __restrict__ src, _Float16* __restrict__ dst, int n) {
  int i = (blockIdx.x * blockDim.x + threadIdx.x) * 4;
  if (i < n) {
    f4 v = *(const f4*)(src + i);
    h4 o; o[0]=(_Float16)v[0]; o[1]=(_Float16)v[1]; o[2]=(_Float16)v[2]; o[3]=(_Float16)v[3];
    *(h4*)(dst + i) = o;
  }
}

__global__ void k_transp(const float* __restrict__ W, _Float16* __restrict__ WT) {
  int n = blockIdx.x * 64 + (threadIdx.x & 63);
  int k = blockIdx.y * 4  + (threadIdx.x >> 6);
  WT[(size_t)n * D_ + k] = (_Float16)W[(size_t)k * D_ + n];
}

// pack mask to 1 bit/element: Mp[dword w of row] bit j = (M[row, w*32+j] != 0)
__global__ void k_packmask(const int* __restrict__ M, unsigned* __restrict__ Mp) {
  int idx = blockIdx.x * blockDim.x + threadIdx.x;     // one thread per element
  unsigned long long bal = __ballot(M[idx] != 0);      // wave32 -> low 32 bits
  if ((threadIdx.x & 31) == 0) Mp[idx >> 5] = (unsigned)bal;
}

// ---------------- fused QKV projection ----------------
// block = 256 (8 waves), one 64-col (= one head) panel of the weight staged in LDS.
// z=0: Qp(f32)+Qa(f16); z=1: Ka(f16); z=2: VT(f16, [b][h][64][Nk])
__global__ void k_proj(const _Float16* __restrict__ Qh, const _Float16* __restrict__ Kh,
                       const _Float16* __restrict__ WqT, const _Float16* __restrict__ WkT,
                       const _Float16* __restrict__ WvT,
                       const float* __restrict__ bq, const float* __restrict__ bk,
                       const float* __restrict__ bv,
                       float* __restrict__ Qp, _Float16* __restrict__ Qa,
                       _Float16* __restrict__ Ka, _Float16* __restrict__ VT) {
  __shared__ __align__(16) _Float16 Wt[64 * D_];   // 64 KB weight panel
  int h = blockIdx.y, z = blockIdx.z, tid = threadIdx.x;
  const _Float16* X    = (z == 0) ? Qh : Kh;
  const _Float16* WT_  = (z == 0) ? WqT : (z == 1) ? WkT : WvT;
  const float*    bias = (z == 0) ? bq  : (z == 1) ? bk  : bv;

  {
    const char* g = (const char*)(WT_ + (size_t)(h * 64) * D_) + tid * 16;
    unsigned    s = lds_off(&Wt[0]) + tid * 16;
#pragma unroll
    for (int i = 0; i < 16; ++i) async_b128(s + i * 4096, g + i * 4096);
    asm volatile("s_wait_asynccnt 0" ::: "memory");
  }
  __syncthreads();

  int mt = blockIdx.x * 8 + (tid >> 5);
  const _Float16* xrow = X + (size_t)mt * 16 * D_;
  v8f acc[4] = {v8f{}, v8f{}, v8f{}, v8f{}};
  v16h a = load_fragA(xrow, D_);
  for (int k = 0; k < D_; k += 32) {
    v16h an = a;
    if (k + 32 < D_) an = load_fragA(xrow + k + 32, D_);   // prefetch next A
    v16h bw[4];
#pragma unroll
    for (int t = 0; t < 4; ++t) bw[t] = load_fragB(&Wt[(size_t)(t * 16) * D_ + k], D_);
#pragma unroll
    for (int t = 0; t < 4; ++t) acc[t] = wmma(a, bw[t], acc[t]);
    a = an;
  }

  int l = laneid(), nn = l & 15, mo = (l >> 4) * 8;
#pragma unroll
  for (int t = 0; t < 4; ++t) {
    int n = h * 64 + t * 16 + nn;
    float bb = bias[n];
    if (z == 2) {
      h8 pk;
#pragma unroll
      for (int r = 0; r < 8; ++r) pk[r] = (_Float16)(acc[t][r] + bb);
      int b   = mt >> 6;
      int tok = ((mt & 63) << 4) + mo;
      *(h8*)(VT + ((size_t)(b * H_ + h) * DH_ + t * 16 + nn) * N_ + tok) = pk;
    } else {
#pragma unroll
      for (int r = 0; r < 8; ++r) {
        float v = acc[t][r] + bb;
        size_t idx = ((size_t)mt * 16 + r + mo) * D_ + n;
        if (z == 0) { Qp[idx] = v; Qa[idx] = (_Float16)v; }
        else        { Ka[idx] = (_Float16)v; }
      }
    }
  }
}

// ---------------- flash attention (one wave per b,h,16 q-rows) ----------------
// K/V tiles double-buffered in LDS via async copies; P staged through LDS; bit-packed mask.
__global__ void k_attn(const _Float16* __restrict__ Qa, const _Float16* __restrict__ Ka,
                       const _Float16* __restrict__ VT, const float* __restrict__ Qp,
                       const unsigned* __restrict__ Mp, float* __restrict__ Og) {
  __shared__ __align__(16) _Float16 Kt[2][32 * 64];   // 32 tokens x 64 ch
  __shared__ __align__(16) _Float16 Vt[2][64 * 32];   // 64 ch x 32 tokens
  __shared__ __align__(16) _Float16 Pt[16 * 32];
  int b = blockIdx.x >> 6, qt = blockIdx.x & 63, h = blockIdx.y;
  int l = laneid(), nn = l & 15, mo = (l >> 4) * 8;

  const _Float16* qptr = Qa + ((size_t)(b * N_) + qt * 16) * D_ + h * DH_;
  v16h qf0 = load_fragA(qptr,      D_);
  v16h qf1 = load_fragA(qptr + 32, D_);

  // per-lane async-copy bases: K rows = tokens (1024 B apart), V rows = channels (2048 B)
  const char* gK = (const char*)(Ka + (size_t)(b * N_) * D_ + h * DH_)
                   + (l >> 3) * (D_ * 2) + (l & 7) * 16;
  const char* gV = (const char*)(VT + (size_t)(b * H_ + h) * DH_ * N_)
                   + (l >> 2) * (N_ * 2) + (l & 3) * 16;
  unsigned lK0 = lds_off(&Kt[0][0]) + (l >> 3) * 128 + (l & 7) * 16;
  unsigned lK1 = lds_off(&Kt[1][0]) + (l >> 3) * 128 + (l & 7) * 16;
  unsigned lV0 = lds_off(&Vt[0][0]) + (l >> 2) * 64 + (l & 3) * 16;
  unsigned lV1 = lds_off(&Vt[1][0]) + (l >> 2) * 64 + (l & 3) * 16;

  auto issue = [&](int buf, int kb) {
    unsigned lk = buf ? lK1 : lK0, lv = buf ? lV1 : lV0;
    const char* gk = gK + (size_t)kb * (D_ * 2);
    const char* gv = gV + (size_t)kb * 2;
#pragma unroll
    for (int i = 0; i < 8; ++i) async_b128(lk + i * 512, gk + i * 4096);          // 4 tok rows/instr
#pragma unroll
    for (int i = 0; i < 8; ++i) async_b128(lv + i * 512, gv + (size_t)i * 16384); // 8 ch rows/instr
  };

  const unsigned* mp = Mp + ((size_t)b * N_ + (size_t)qt * 16) * (N_ / 32);

  v8f o[4] = {v8f{}, v8f{}, v8f{}, v8f{}};
  float rmax[8], rsum[8];
#pragma unroll
  for (int r = 0; r < 8; ++r) { rmax[r] = -1e30f; rsum[r] = 0.f; }

  issue(0, 0);
  int buf = 0;
  for (int kb = 0; kb < N_; kb += 32, buf ^= 1) {
    if (kb + 32 < N_) {
      issue(buf ^ 1, kb + 32);                         // prefetch next tile
      asm volatile("s_wait_asynccnt 16" ::: "memory"); // current tile done (in-order)
    } else {
      asm volatile("s_wait_asynccnt 0" ::: "memory");
    }
    const _Float16* kt = &Kt[buf][0];
    const _Float16* vt = &Vt[buf][0];

    v16h kf0 = load_fragB(kt,                64);
    v16h kf1 = load_fragB(kt + 32,           64);
    v16h kf2 = load_fragB(kt + 16 * 64,      64);
    v16h kf3 = load_fragB(kt + 16 * 64 + 32, 64);
    v8f s[2];
    { v8f z = v8f{}; z = wmma(qf0, kf0, z); s[0] = wmma(qf1, kf1, z); }
    { v8f z = v8f{}; z = wmma(qf0, kf2, z); s[1] = wmma(qf1, kf3, z); }

    int w = kb >> 5;
#pragma unroll
    for (int r = 0; r < 8; ++r) {
      int row = r + mo;
      unsigned mw = mp[row * (N_ / 32) + w];           // 32 mask bits for this k-window
      float s0 = s[0][r] * SCALE; if (!((mw >> nn) & 1u))        s0 = -1e30f;
      float s1 = s[1][r] * SCALE; if (!((mw >> (nn + 16)) & 1u)) s1 = -1e30f;
      float mx = fmaxf(s0, s1);
#pragma unroll
      for (int d = 1; d < 16; d <<= 1) mx = fmaxf(mx, __shfl_xor(mx, d, 32));
      float mnew = fmaxf(rmax[r], mx);
      float corr = __expf(rmax[r] - mnew);
      rmax[r] = mnew;
      float p0 = __expf(s0 - mnew);
      float p1 = __expf(s1 - mnew);
      float ps = p0 + p1;
#pragma unroll
      for (int d = 1; d < 16; d <<= 1) ps += __shfl_xor(ps, d, 32);
      rsum[r] = rsum[r] * corr + ps;
#pragma unroll
      for (int t = 0; t < 4; ++t) o[t][r] *= corr;
      Pt[row * 32 + nn]      = (_Float16)p0;
      Pt[row * 32 + 16 + nn] = (_Float16)p1;
    }
    asm volatile("s_wait_dscnt 0" ::: "memory");
    v16h pf = load_fragA((const _Float16*)Pt, 32);
    v16h vb[4];
#pragma unroll
    for (int t = 0; t < 4; ++t) vb[t] = load_fragB(vt + (size_t)(t * 16) * 32, 32);
#pragma unroll
    for (int t = 0; t < 4; ++t) o[t] = wmma(pf, vb[t], o[t]);
    asm volatile("" ::: "memory");
  }
#pragma unroll
  for (int t = 0; t < 4; ++t)
#pragma unroll
    for (int r = 0; r < 8; ++r) {
      size_t idx = ((size_t)(b * N_) + qt * 16 + r + mo) * D_ + h * DH_ + t * 16 + nn;
      Og[idx] = Qp[idx] + o[t][r] / rsum[r];
    }
}

// ---------------- layernorm (one wave per 512-elem row) ----------------
__global__ void k_ln(const float* __restrict__ X, const float* __restrict__ g,
                     const float* __restrict__ bt, float* __restrict__ Y32,
                     _Float16* __restrict__ Y16) {
  int row = blockIdx.x * 8 + (threadIdx.x >> 5);
  int l = laneid();
  const float* x = X + (size_t)row * D_;
  f4 v[4]; float s = 0.f, s2 = 0.f;
#pragma unroll
  for (int i = 0; i < 4; ++i) {
    v[i] = *(const f4*)(x + l * 16 + i * 4);
#pragma unroll
    for (int j = 0; j < 4; ++j) { s += v[i][j]; s2 += v[i][j] * v[i][j]; }
  }
#pragma unroll
  for (int d = 1; d < 32; d <<= 1) { s += __shfl_xor(s, d, 32); s2 += __shfl_xor(s2, d, 32); }
  float mean = s * (1.f / D_);
  float var  = s2 * (1.f / D_) - mean * mean;
  float rstd = rsqrtf(var + 1e-5f);
#pragma unroll
  for (int i = 0; i < 4; ++i)
#pragma unroll
    for (int j = 0; j < 4; ++j) {
      int c = l * 16 + i * 4 + j;
      float y = (v[i][j] - mean) * rstd * g[c] + bt[c];
      Y32[(size_t)row * D_ + c] = y;
      if (Y16) Y16[(size_t)row * D_ + c] = (_Float16)y;
    }
}

// ---------------- FFN: Y = X1 + relu(X1 @ Wo + bo) ----------------
__global__ void k_ffn(const _Float16* __restrict__ X1h, const _Float16* __restrict__ WoT,
                      const float* __restrict__ bo, const float* __restrict__ X1,
                      float* __restrict__ Y) {
  __shared__ __align__(16) _Float16 Wt[64 * D_];
  int h = blockIdx.y, tid = threadIdx.x;
  {
    const char* g = (const char*)(WoT + (size_t)(h * 64) * D_) + tid * 16;
    unsigned    s = lds_off(&Wt[0]) + tid * 16;
#pragma unroll
    for (int i = 0; i < 16; ++i) async_b128(s + i * 4096, g + i * 4096);
    asm volatile("s_wait_asynccnt 0" ::: "memory");
  }
  __syncthreads();

  int mt = blockIdx.x * 8 + (tid >> 5);
  const _Float16* xrow = X1h + (size_t)mt * 16 * D_;
  v8f acc[4] = {v8f{}, v8f{}, v8f{}, v8f{}};
  v16h a = load_fragA(xrow, D_);
  for (int k = 0; k < D_; k += 32) {
    v16h an = a;
    if (k + 32 < D_) an = load_fragA(xrow + k + 32, D_);
    v16h bw[4];
#pragma unroll
    for (int t = 0; t < 4; ++t) bw[t] = load_fragB(&Wt[(size_t)(t * 16) * D_ + k], D_);
#pragma unroll
    for (int t = 0; t < 4; ++t) acc[t] = wmma(a, bw[t], acc[t]);
    a = an;
  }
  int l = laneid(), nn = l & 15, mo = (l >> 4) * 8;
#pragma unroll
  for (int t = 0; t < 4; ++t) {
    int n = h * 64 + t * 16 + nn;
    float bb = bo[n];
#pragma unroll
    for (int r = 0; r < 8; ++r) {
      size_t idx = ((size_t)mt * 16 + r + mo) * D_ + n;
      float v = acc[t][r] + bb;
      Y[idx] = X1[idx] + fmaxf(v, 0.f);
    }
  }
}

extern "C" void kernel_launch(void* const* d_in, const int* in_sizes, int n_in,
                              void* d_out, int out_size, void* d_ws, size_t ws_size,
                              hipStream_t stream) {
  const float* Q  = (const float*)d_in[0];
  const float* K  = (const float*)d_in[1];
  const float* Wq = (const float*)d_in[2];
  const float* bq = (const float*)d_in[3];
  const float* Wk = (const float*)d_in[4];
  const float* bk = (const float*)d_in[5];
  const float* Wv = (const float*)d_in[6];
  const float* bv = (const float*)d_in[7];
  const float* Wo = (const float*)d_in[8];
  const float* bo = (const float*)d_in[9];
  const float* g0 = (const float*)d_in[10];
  const float* b0 = (const float*)d_in[11];
  const float* g1 = (const float*)d_in[12];
  const float* b1 = (const float*)d_in[13];
  const int*   M  = (const int*)d_in[14];
  float* out = (float*)d_out;

  const size_t NE = (size_t)B_ * N_ * D_;
  const size_t NM = (size_t)B_ * N_ * N_;   // mask elements
  size_t off = 0;
  auto ws = [&](size_t bytes) { void* p = (char*)d_ws + off; off += (bytes + 255) & ~(size_t)255; return p; };
  _Float16* Qh  = (_Float16*)ws(NE * 2);
  _Float16* Kh  = (_Float16*)ws(NE * 2);
  _Float16* WqT = (_Float16*)ws((size_t)D_ * D_ * 2);
  _Float16* WkT = (_Float16*)ws((size_t)D_ * D_ * 2);
  _Float16* WvT = (_Float16*)ws((size_t)D_ * D_ * 2);
  _Float16* WoT = (_Float16*)ws((size_t)D_ * D_ * 2);
  float*    Qp  = (float*)ws(NE * 4);
  _Float16* Qa  = (_Float16*)ws(NE * 2);
  _Float16* Ka  = (_Float16*)ws(NE * 2);
  _Float16* VT  = (_Float16*)ws(NE * 2);
  float*    Og  = (float*)ws(NE * 4);
  float*    X1  = (float*)ws(NE * 4);
  _Float16* X1h = (_Float16*)ws(NE * 2);
  float*    Y   = (float*)ws(NE * 4);
  unsigned* Mp  = (unsigned*)ws(NM / 8);    // 1 bit per mask element
  (void)ws_size;

  k_cvt16<<<dim3((NE / 4 + 255) / 256), 256, 0, stream>>>(Q, Qh, (int)NE);
  k_cvt16<<<dim3((NE / 4 + 255) / 256), 256, 0, stream>>>(K, Kh, (int)NE);
  k_transp<<<dim3(8, 128), 256, 0, stream>>>(Wq, WqT);
  k_transp<<<dim3(8, 128), 256, 0, stream>>>(Wk, WkT);
  k_transp<<<dim3(8, 128), 256, 0, stream>>>(Wv, WvT);
  k_transp<<<dim3(8, 128), 256, 0, stream>>>(Wo, WoT);
  k_packmask<<<dim3(NM / 256), 256, 0, stream>>>(M, Mp);

  k_proj<<<dim3(64, 8, 3), 256, 0, stream>>>(Qh, Kh, WqT, WkT, WvT, bq, bk, bv,
                                             Qp, Qa, Ka, VT);
  k_attn<<<dim3(512, 8), 32, 0, stream>>>(Qa, Ka, VT, Qp, Mp, Og);
  k_ln<<<dim3(1024), 256, 0, stream>>>(Og, g0, b0, X1, X1h);
  k_ffn<<<dim3(64, 8), 256, 0, stream>>>(X1h, WoT, bo, X1, Y);
  k_ln<<<dim3(1024), 256, 0, stream>>>(Y, g1, b1, out, (_Float16*)nullptr);
}
